// GNNVulnerabilityDetector_84550726189262
// MI455X (gfx1250) — matrix operand order repaired and checked by reading, not compiled, not run on hardware
//
#include <hip/hip_runtime.h>
#include <hip/hip_bf16.h>

typedef __attribute__((ext_vector_type(16))) _Float16 v16h;
typedef __attribute__((ext_vector_type(8)))  float    v8f;

#define HID 64
#define NGRAPH 256

// ---------------- small helpers ----------------
__device__ __forceinline__ unsigned f32_order_enc(float x) {
    unsigned u = __float_as_uint(x);
    return (u & 0x80000000u) ? ~u : (u | 0x80000000u);
}
__device__ __forceinline__ float f32_order_dec(unsigned u) {
    unsigned b = (u & 0x80000000u) ? (u & 0x7FFFFFFFu) : ~u;
    return __uint_as_float(b);
}

// ---------------- init / degree ----------------
__global__ void k_zero_f32(float* __restrict__ p, long long n) {
    long long i = (long long)blockIdx.x * blockDim.x + threadIdx.x;
    if (i < n) p[i] = 0.0f;
}

__global__ void k_init_deg(float* __restrict__ deg, int n) {
    int i = blockIdx.x * blockDim.x + threadIdx.x;
    if (i < n) deg[i] = 1.0f;          // self loop
}

__global__ void k_deg_edges(float* __restrict__ deg, const int* __restrict__ dst, int E) {
    int e = blockIdx.x * blockDim.x + threadIdx.x;
    if (e < E) atomicAdd(&deg[dst[e]], 1.0f);
}

__global__ void k_rsqrt(float* __restrict__ dinv, int n) {
    int i = blockIdx.x * blockDim.x + threadIdx.x;
    if (i < n) {
        float d = dinv[i];
        dinv[i] = (d > 0.0f) ? rsqrtf(d) : 0.0f;
    }
}

// ---------------- weight pre-pack: fp32 [K,64] -> f16 WMMA B-fragments ----------------
// Fragment f = (ks)*4 + t holds lane-major data: P[((f*32)+lane)*16 + i] =
// (f16) W[(ks*32 + (lane>>4)*16 + i) * 64 + t*16 + (lane&15)]
__global__ void k_pack_w(const float* __restrict__ W, _Float16* __restrict__ P, int K) {
    int gid = blockIdx.x * blockDim.x + threadIdx.x;
    int total = (K >> 5) * 2048;                 // (K/32) k-steps * 4 tiles * 32 lanes * 16 halves
    if (gid >= total) return;
    int i    =  gid        & 15;
    int lane = (gid >> 4)  & 31;
    int t    = (gid >> 9)  & 3;
    int ks   =  gid >> 11;
    int k    = ks * 32 + (lane >> 4) * 16 + i;
    int col  = t * 16 + (lane & 15);
    P[gid] = (_Float16)W[(size_t)k * HID + col];
}

// ---------------- WMMA GEMM: out[M,64] = A[M,K] @ W[K,64] (+bias, relu) ----------------
// One wave handles a 16-row tile and all four 16-col tiles. K multiple of 32.
// B comes pre-packed in fragment layout (contiguous 32B per lane per fragment).
__global__ __launch_bounds__(128)
void k_gemm_wmma(const float* __restrict__ A, const _Float16* __restrict__ Wp,
                 float* __restrict__ out, int M, int K,
                 const float* __restrict__ bias, int do_relu)
{
    int mtiles = (M + 15) >> 4;
    int wave = blockIdx.x * (blockDim.x >> 5) + (threadIdx.x >> 5);
    if (wave >= mtiles) return;                 // wave-uniform exit: EXEC stays all-1s

    int lane = threadIdx.x & 31;
    int h = lane >> 4;                          // lane half
    int n = lane & 15;                          // row (A) / column (B,C,D) within tile
    int row0 = wave << 4;

    int arow = row0 + n;
    if (arow >= M) arow = M - 1;                // clamp loads; stores guarded below
    const float* Ap = A + (size_t)arow * K;
    const v16h* Bp = (const v16h*)Wp;           // fragment index = ((ks/32)*4 + t)*32 + lane

    v8f acc0 = {}, acc1 = {}, acc2 = {}, acc3 = {};

    for (int ks = 0; ks < K; ks += 32) {
        // A fragment: one contiguous 16-float (64B) run at k = 16h, converted fp32 -> f16
        v16h a;
#pragma unroll
        for (int i = 0; i < 8; ++i) a[i]     = (_Float16)Ap[ks + h * 16 + i];
#pragma unroll
        for (int i = 0; i < 8; ++i) a[8 + i] = (_Float16)Ap[ks + h * 16 + 8 + i];

        // B fragments: pre-packed, lane-contiguous 32B vector loads
        int base = (ks >> 5) * 4 * 32 + lane;
        v16h b0 = Bp[base +  0];
        v16h b1 = Bp[base + 32];
        v16h b2 = Bp[base + 64];
        v16h b3 = Bp[base + 96];

        acc0 = __builtin_amdgcn_wmma_f32_16x16x32_f16(false, a, false, b0, (short)0, acc0, false, false);
        acc1 = __builtin_amdgcn_wmma_f32_16x16x32_f16(false, a, false, b1, (short)0, acc1, false, false);
        acc2 = __builtin_amdgcn_wmma_f32_16x16x32_f16(false, a, false, b2, (short)0, acc2, false, false);
        acc3 = __builtin_amdgcn_wmma_f32_16x16x32_f16(false, a, false, b3, (short)0, acc3, false, false);
    }

#pragma unroll
    for (int v = 0; v < 8; ++v) {
        int r = row0 + v + 8 * h;
        if (r < M) {
            float vals[4] = { acc0[v], acc1[v], acc2[v], acc3[v] };
#pragma unroll
            for (int t = 0; t < 4; ++t) {
                int col = t * 16 + n;
                float o = vals[t];
                if (bias) o += bias[col];
                if (do_relu) o = fmaxf(o, 0.0f);
                out[(size_t)r * HID + col] = o;
            }
        }
    }
}

// ---------------- edge scatter: agg[dst] += h[src] * dinv[src]*dinv[dst] ----------------
// One thread per (edge, 4-feature chunk): 16-byte gather + 4 f32 atomics (L2-resident).
__global__ void k_scatter(const float* __restrict__ h, const float* __restrict__ dinv,
                          const int* __restrict__ src, const int* __restrict__ dst,
                          float* __restrict__ agg, int E)
{
    long long gid = (long long)blockIdx.x * blockDim.x + threadIdx.x;
    long long total = (long long)E * 16;
    if (gid >= total) return;
    int e = (int)(gid >> 4);
    int c = (int)(gid & 15) * 4;
    int s = src[e], d = dst[e];
    float w = dinv[s] * dinv[d];
    const float4 hv = *(const float4*)(h + (size_t)s * HID + c);
    float* ap = agg + (size_t)d * HID + c;
    atomicAdd(ap + 0, hv.x * w);
    atomicAdd(ap + 1, hv.y * w);
    atomicAdd(ap + 2, hv.z * w);
    atomicAdd(ap + 3, hv.w * w);
}

// ---------------- fused self-loop + bias (+relu) ----------------
__global__ void k_finalize(float* __restrict__ agg, const float* __restrict__ h,
                           const float* __restrict__ dinv, const float* __restrict__ bias,
                           int do_relu, int N)
{
    long long gid = (long long)blockIdx.x * blockDim.x + threadIdx.x;
    if (gid >= (long long)N * HID) return;
    int i = (int)(gid >> 6);
    int c = (int)(gid & 63);
    float di = dinv[i];
    float v = agg[gid] + h[gid] * di * di + bias[c];
    if (do_relu) v = fmaxf(v, 0.0f);
    agg[gid] = v;
}

// ---------------- segment mean/max pooling ----------------
__global__ void k_pool(const float* __restrict__ h, const int* __restrict__ batch,
                       float* __restrict__ cnt, float* __restrict__ psum,
                       unsigned* __restrict__ pmax, int N)
{
    long long gid = (long long)blockIdx.x * blockDim.x + threadIdx.x;
    if (gid >= (long long)N * HID) return;
    int i = (int)(gid >> 6);
    int f = (int)(gid & 63);
    int g = batch[i];
    float v = h[gid];
    atomicAdd(&psum[(size_t)g * HID + f], v);
    atomicMax(&pmax[(size_t)g * HID + f], f32_order_enc(v));
    if (f == 0) atomicAdd(&cnt[g], 1.0f);
}

__global__ void k_pool_finalize(const float* __restrict__ cnt, const float* __restrict__ psum,
                                const unsigned* __restrict__ pmax, float* __restrict__ gbuf)
{
    int gid = blockIdx.x * blockDim.x + threadIdx.x;
    if (gid >= NGRAPH * 2 * HID) return;
    int g = gid >> 7;
    int f = gid & 127;
    float c = cnt[g];
    float o;
    if (f < HID) {
        o = psum[g * HID + f] / fmaxf(c, 1.0f);
    } else {
        o = (c > 0.0f) ? f32_order_dec(pmax[g * HID + (f - HID)]) : 0.0f;
    }
    gbuf[gid] = o;
}

// ---------------- logits: out[256,2] = hcls @ Wc2 + bc2 ----------------
__global__ void k_logits(const float* __restrict__ hcls, const float* __restrict__ Wc2,
                         const float* __restrict__ bc2, float* __restrict__ out)
{
    int gid = blockIdx.x * blockDim.x + threadIdx.x;
    if (gid >= NGRAPH * 2) return;
    int g = gid >> 1;
    int o = gid & 1;
    float s = bc2[o];
#pragma unroll
    for (int k = 0; k < HID; ++k) s += hcls[g * HID + k] * Wc2[k * 2 + o];
    out[gid] = s;
}

// ---------------- launch ----------------
static inline unsigned cdiv_ll(long long a, long long b) { return (unsigned)((a + b - 1) / b); }

extern "C" void kernel_launch(void* const* d_in, const int* in_sizes, int n_in,
                              void* d_out, int out_size, void* d_ws, size_t ws_size,
                              hipStream_t stream)
{
    const float* x   = (const float*)d_in[0];
    const int*   ei  = (const int*)d_in[1];      // [2, E]
    const int*   bat = (const int*)d_in[2];
    const float* W0  = (const float*)d_in[3];
    const float* b0  = (const float*)d_in[4];
    const float* W1  = (const float*)d_in[5];
    const float* b1  = (const float*)d_in[6];
    const float* W2  = (const float*)d_in[7];
    const float* b2  = (const float*)d_in[8];
    const float* Wc1 = (const float*)d_in[9];
    const float* bc1 = (const float*)d_in[10];
    const float* Wc2 = (const float*)d_in[11];
    const float* bc2 = (const float*)d_in[12];

    const int N = in_sizes[0] / 128;             // 100000
    const int E = in_sizes[1] / 2;               // 3200000
    const int* src = ei;
    const int* dst = ei + E;

    float* ws   = (float*)d_ws;
    float* bufA = ws;                 ws += (size_t)N * HID;   // GEMM outputs (pre-agg)
    float* bufB = ws;                 ws += (size_t)N * HID;   // aggregation / activations
    float* dinv = ws;                 ws += N;
    float* cnt  = ws;                 ws += NGRAPH;
    float* psum = ws;                 ws += NGRAPH * HID;
    unsigned* pmax = (unsigned*)ws;   ws += NGRAPH * HID;
    float* gbuf = ws;                 ws += NGRAPH * 2 * HID;
    float* hcls = ws;                 ws += NGRAPH * HID;
    _Float16* pW0  = (_Float16*)ws;   ws += 4096;              // 8192 f16 (K=128)
    _Float16* pW1  = (_Float16*)ws;   ws += 2048;              // 4096 f16 (K=64)
    _Float16* pW2  = (_Float16*)ws;   ws += 2048;
    _Float16* pWc1 = (_Float16*)ws;   ws += 4096;

    const int T = 256;
    const long long NH = (long long)N * HID;
    const long long E16 = (long long)E * 16;

    // One-shot weight packs (fp32 -> f16 WMMA B-fragments)
    k_pack_w<<<cdiv_ll(8192, T), T, 0, stream>>>(W0,  pW0,  128);
    k_pack_w<<<cdiv_ll(4096, T), T, 0, stream>>>(W1,  pW1,  64);
    k_pack_w<<<cdiv_ll(4096, T), T, 0, stream>>>(W2,  pW2,  64);
    k_pack_w<<<cdiv_ll(8192, T), T, 0, stream>>>(Wc1, pWc1, 128);

    // GCN normalization: deg -> dinv
    k_init_deg<<<cdiv_ll(N, T), T, 0, stream>>>(dinv, N);
    k_deg_edges<<<cdiv_ll(E, T), T, 0, stream>>>(dinv, dst, E);
    k_rsqrt<<<cdiv_ll(N, T), T, 0, stream>>>(dinv, N);

    auto gemm = [&](const float* A, const _Float16* Wp, float* out, int M, int K,
                    const float* bias, int relu) {
        int mtiles = (M + 15) >> 4;
        k_gemm_wmma<<<cdiv_ll(mtiles, 4), 128, 0, stream>>>(A, Wp, out, M, K, bias, relu);
    };
    auto layer_agg = [&](const float* bias, int relu) {
        k_zero_f32<<<cdiv_ll(NH, T), T, 0, stream>>>(bufB, NH);
        k_scatter<<<cdiv_ll(E16, T), T, 0, stream>>>(bufA, dinv, src, dst, bufB, E);
        k_finalize<<<cdiv_ll(NH, T), T, 0, stream>>>(bufB, bufA, dinv, bias, relu, N);
    };

    // Layer 0: h = relu(agg(x @ W0) + b0)
    gemm(x, pW0, bufA, N, 128, nullptr, 0);
    layer_agg(b0, 1);
    // Layer 1
    gemm(bufB, pW1, bufA, N, HID, nullptr, 0);
    layer_agg(b1, 1);
    // Layer 2 (no relu)
    gemm(bufB, pW2, bufA, N, HID, nullptr, 0);
    layer_agg(b2, 0);

    // Pooling (mean + max per graph)
    k_zero_f32<<<cdiv_ll(NGRAPH, T), T, 0, stream>>>(cnt, NGRAPH);
    k_zero_f32<<<cdiv_ll(NGRAPH * HID, T), T, 0, stream>>>(psum, NGRAPH * HID);
    k_zero_f32<<<cdiv_ll(NGRAPH * HID, T), T, 0, stream>>>((float*)pmax, NGRAPH * HID); // enc 0 < enc(-inf)
    k_pool<<<cdiv_ll(NH, T), T, 0, stream>>>(bufB, bat, cnt, psum, pmax, N);
    k_pool_finalize<<<cdiv_ll(NGRAPH * 2 * HID, T), T, 0, stream>>>(cnt, psum, pmax, gbuf);

    // Classifier head: hcls = relu(g @ Wc1 + bc1)  (WMMA, M=256, K=128)
    gemm(gbuf, pWc1, hcls, NGRAPH, 2 * HID, bc1, 1);
    // logits = hcls @ Wc2 + bc2
    k_logits<<<cdiv_ll(NGRAPH * 2, T), T, 0, stream>>>(hcls, Wc2, bc2, (float*)d_out);
}
